// PrototypeContextAttention_69312182223285
// MI455X (gfx1250) — compile-verified, matrix operand order checked by reading, not run on hardware
//
#include <hip/hip_runtime.h>
#include <stdint.h>

// ---------------------------------------------------------------------------
// PrototypeContextAttention for MI455X (gfx1250, wave32, WMMA + TDM)
// B=8192, P=4096, D=1024, heads=16, dh=64, topk=6
// ---------------------------------------------------------------------------

typedef __attribute__((ext_vector_type(16))) __bf16 v16bf;
typedef __attribute__((ext_vector_type(8)))  __bf16 v8bf;
typedef __attribute__((ext_vector_type(8)))  float  v8f;

#define TOPK   6
#define NHEADS 16
#define DHEAD  64

// Tensor Data Mover availability (toolchain-dependent; compile-safe guards)
#if defined(__has_builtin)
#if __has_builtin(__builtin_amdgcn_tensor_load_to_lds) && \
    __has_builtin(__builtin_amdgcn_s_wait_tensorcnt)
#define USE_TDM 1
#endif
#endif
#ifndef USE_TDM
#define USE_TDM 0
#endif

// ---------------------------------------------------------------------------
// f32 -> bf16 (round to nearest even)
// ---------------------------------------------------------------------------
__device__ __forceinline__ unsigned short f32_to_bf16_rne(float f) {
  union { float f; unsigned int u; } x;
  x.f = f;
  unsigned int u = x.u;
  u += 0x7FFFu + ((u >> 16) & 1u);
  return (unsigned short)(u >> 16);
}

__global__ __launch_bounds__(256) void f32_to_bf16_kernel(
    const float* __restrict__ src, unsigned short* __restrict__ dst, int n) {
  int i = blockIdx.x * blockDim.x + threadIdx.x;
  int stride = gridDim.x * blockDim.x;
  for (; i < n; i += stride) dst[i] = f32_to_bf16_rne(src[i]);
}

// ---------------------------------------------------------------------------
// TDM: DMA a 128-row x 32-col bf16 tile of W[N,K] into LDS (padded rows).
// D# descriptor per CDNA5 ISA ch.8: group0 = {count/lds_addr/global_addr/type},
// group1 = {flags, tensor dims, tile dims, strides}. LDS padding: 16B after
// every 64B row -> 80B row stride (bank-conflict-free ds_load_b128 later).
// ---------------------------------------------------------------------------
#define B_LDS_STRIDE 80   // bytes per padded LDS row (64B data + 16B pad)

#if USE_TDM
typedef unsigned int u32x4 __attribute__((ext_vector_type(4)));
typedef int          i32x4 __attribute__((ext_vector_type(4)));
typedef int          i32x8 __attribute__((ext_vector_type(8)));

__device__ __forceinline__ void tdm_load_wtile(const __bf16* gsrc,
                                               unsigned lds_byte_off,
                                               int K, int N) {
  const unsigned long long ga = (unsigned long long)(uintptr_t)gsrc;
  u32x4 g0;
  g0.x = 1u;                                                 // count=1 (valid user D#)
  g0.y = lds_byte_off;                                       // lds_addr
  g0.z = (unsigned)(ga & 0xFFFFFFFFu);                       // global_addr[31:0]
  g0.w = (unsigned)((ga >> 32) & 0x01FFFFFFu) | (2u << 30);  // addr[56:32] | type=2

  i32x8 g1;
  g1[0] = (int)((1u << 16)        // data_size = 2 bytes (bf16)
              | (1u << 20)        // pad_enable
              | (3u << 22)        // pad_interval: every 64B
              | (3u << 25));      // pad_amount: 4 DWORDs = 16B
  g1[1] = (int)(((unsigned)K & 0xFFFFu) << 16);                       // tensor_dim0 lo
  g1[2] = (int)(((unsigned)K >> 16) | (((unsigned)N & 0xFFFFu) << 16)); // dim0 hi | dim1 lo
  g1[3] = (int)(((unsigned)N >> 16) | (32u << 16));                   // dim1 hi | tile_dim0=32
  g1[4] = 128;                                                        // tile_dim1 = 128 rows
  g1[5] = K;                                                          // dim0_stride[31:0]
  g1[6] = 0;                                                          // stride hi / dim1_stride
  g1[7] = 0;

  i32x4 z4 = {0, 0, 0, 0};
#if __clang_major__ >= 23
  i32x8 z8 = {0, 0, 0, 0, 0, 0, 0, 0};
  __builtin_amdgcn_tensor_load_to_lds(g0, g1, z4, z4, z8, 0);
#else
  __builtin_amdgcn_tensor_load_to_lds(g0, g1, z4, z4, 0);
#endif
}
#endif  // USE_TDM

// ---------------------------------------------------------------------------
// WMMA bf16 GEMM:  C[M,N] = A[M,K] @ W[N,K]^T + bias
// Block = 256 threads = 8 waves -> 128x128 tile; wave = 32x64 (2x4 WMMA accs).
// A-tiles: direct global_load_b128 (software pipelined by compiler).
// B-tiles (weights, 4-way shared across M-waves): TDM DMA into double-buffered
// LDS, one tensor_load_to_lds per K-step, consumed via ds_load_b128.
// Per-lane WMMA layouts per ISA 7.12.2:
//   A/B fragment: lane L -> m/n = L&15, K-half = 8*(L>>4).
//   C/D fragment: lane L, vgpr r -> m = r + 8*(L>>4), n = L&15.
// ---------------------------------------------------------------------------
__global__ __launch_bounds__(256) void gemm_bf16_wmma(
    const __bf16* __restrict__ A,        // [M,K]
    const __bf16* __restrict__ W,        // [N,K]
    const float*  __restrict__ bias,     // [N] or nullptr
    float*          __restrict__ Cf,     // fp32 out (or nullptr)
    unsigned short* __restrict__ Cb,     // bf16 out (or nullptr)
    int M, int N, int K) {
  __shared__ __align__(16) unsigned char smemB[2][128 * B_LDS_STRIDE];

  const int tid   = threadIdx.x;
  const int lane  = tid & 31;
  const int wave  = tid >> 5;
  const int waveM = wave & 3;             // 4 waves along M
  const int waveN = wave >> 2;            // 2 waves along N
  const int rowBase = blockIdx.y * 128 + waveM * 32;
  const int colBlk  = blockIdx.x * 128;   // block's W-row (output col) base
  const int mLane = lane & 15;
  const int kHalf = (lane >> 4) << 3;     // 0 or 8

  const __bf16* aRow[2];
  aRow[0] = A + (size_t)(rowBase + mLane) * K + kHalf;
  aRow[1] = A + (size_t)(rowBase + 16 + mLane) * K + kHalf;

  const __bf16* wTile = W + (size_t)colBlk * K;   // 128 x K slab of W

  // LDS read base for this wave's 4 N-subtiles
  const unsigned char* bRd[4];
#pragma unroll
  for (int nt = 0; nt < 4; ++nt) {
    const int r = waveN * 64 + nt * 16 + mLane;   // row within 128-row tile
    bRd[nt] = &smemB[0][0] + r * B_LDS_STRIDE + kHalf * 2;
  }
  const unsigned bufBytes = 128 * B_LDS_STRIDE;

  v8f acc[2][4] = {};

#if USE_TDM
  // ---- prologue: DMA first B tile ----
  if (wave == 0) {
    tdm_load_wtile(wTile, (unsigned)(uintptr_t)&smemB[0][0], K, N);
    __builtin_amdgcn_s_wait_tensorcnt(0);
  }
  __syncthreads();
#endif

  for (int k0 = 0; k0 < K; k0 += 32) {
    const int cur = (k0 >> 5) & 1;

#if USE_TDM
    // DMA next B tile into the other buffer while we compute on this one
    if (wave == 0 && k0 + 32 < K)
      tdm_load_wtile(wTile + k0 + 32, (unsigned)(uintptr_t)&smemB[cur ^ 1][0],
                     K, N);
#else
    // staging fallback: cooperative copy of cols 0-7, 16-23 of the 128x32 tile (padded layout)
    {
      const int r = tid >> 1, h = tid & 1;  // 2 threads per 64B row
      const v8bf src =
          *(const v8bf*)(wTile + (size_t)r * K + k0 + h * 16);
      *(v8bf*)(&smemB[0][0] + r * B_LDS_STRIDE + h * 32) = src;
    }
    __syncthreads();
#endif

    __builtin_prefetch(aRow[0] + k0 + 64, 0, 1);

    v16bf a[2], b[4];
#pragma unroll
    for (int mt = 0; mt < 2; ++mt) {
      v8bf lo = *(const v8bf*)(aRow[mt] + k0);
      v8bf hi = *(const v8bf*)(aRow[mt] + k0 + 16);
      a[mt] = __builtin_shufflevector(lo, hi, 0, 1, 2, 3, 4, 5, 6, 7,
                                      8, 9, 10, 11, 12, 13, 14, 15);
    }
#pragma unroll
    for (int nt = 0; nt < 4; ++nt) {
      const unsigned char* p = bRd[nt] + (USE_TDM ? cur * bufBytes : 0u);
      v8bf lo = *(const v8bf*)(p);
      v8bf hi = *(const v8bf*)(p + 32);
      b[nt] = __builtin_shufflevector(lo, hi, 0, 1, 2, 3, 4, 5, 6, 7,
                                      8, 9, 10, 11, 12, 13, 14, 15);
    }
#pragma unroll
    for (int mt = 0; mt < 2; ++mt)
#pragma unroll
      for (int nt = 0; nt < 4; ++nt)
        acc[mt][nt] = __builtin_amdgcn_wmma_f32_16x16x32_bf16(
            false, a[mt], false, b[nt], (short)0, acc[mt][nt], false, false);

#if USE_TDM
    if (wave == 0 && k0 + 32 < K) __builtin_amdgcn_s_wait_tensorcnt(0);
#endif
    __syncthreads();
  }

  const int mOff  = (lane >> 4) << 3;
  const int nLane = lane & 15;
#pragma unroll
  for (int nt = 0; nt < 4; ++nt) {
    const int n  = colBlk + waveN * 64 + nt * 16 + nLane;
    const float bn = bias ? bias[n] : 0.0f;
#pragma unroll
    for (int mt = 0; mt < 2; ++mt) {
#pragma unroll
      for (int r = 0; r < 8; ++r) {
        const int m = rowBase + mt * 16 + mOff + r;
        const float v = acc[mt][nt][r] + bn;
        if (Cf) Cf[(size_t)m * N + n] = v;
        else    Cb[(size_t)m * N + n] = f32_to_bf16_rne(v);
      }
    }
  }
}

// ---------------------------------------------------------------------------
// Top-6 per row of logits [B, P] (P = 4096 = 256 threads * 16 elems).
// 6 iterative argmax passes; lower index wins ties (matches jax top_k order);
// indices emitted in descending-value order.
// ---------------------------------------------------------------------------
__global__ __launch_bounds__(256) void topk6_kernel(
    const float* __restrict__ logits, int* __restrict__ idx_out, int P) {
  const int row = blockIdx.x;
  const int tid = threadIdx.x;
  const int EPT = 16;                       // P / 256
  const int base = tid * EPT;
  const float* rp = logits + (size_t)row * P;

  float v[EPT];
#pragma unroll
  for (int i = 0; i < EPT; ++i) v[i] = rp[base + i];

  __shared__ float sval[256];
  __shared__ int   sidx[256];
  __shared__ int   swin;

  for (int t = 0; t < TOPK; ++t) {
    float m = -__builtin_inff();
    int   mi = 0x7FFFFFFF;
#pragma unroll
    for (int i = 0; i < EPT; ++i) {
      if (v[i] > m) { m = v[i]; mi = base + i; }
    }
    sval[tid] = m;
    sidx[tid] = mi;
    __syncthreads();
    for (int s = 128; s > 0; s >>= 1) {
      if (tid < s) {
        const float ov = sval[tid + s];
        const int   oi = sidx[tid + s];
        if (ov > sval[tid] || (ov == sval[tid] && oi < sidx[tid])) {
          sval[tid] = ov;
          sidx[tid] = oi;
        }
      }
      __syncthreads();
    }
    if (tid == 0) {
      idx_out[(size_t)row * TOPK + t] = sidx[0];
      swin = sidx[0];
    }
    __syncthreads();
    const int w = swin;
    if (w >= base && w < base + EPT) v[w - base] = -__builtin_inff();
    __syncthreads();
  }
}

// ---------------------------------------------------------------------------
// Per-query attention over 6 gathered prototype rows.
// 256 threads / query: 16 threads per head, 4 dims per thread.
// Writes ctx (bf16, input of final GEMM) and head-averaged attn weights.
// ---------------------------------------------------------------------------
__global__ __launch_bounds__(256) void attn_topk6_kernel(
    const float* __restrict__ q,        // [B, D]
    const float* __restrict__ Kb,       // [P, D]
    const float* __restrict__ Vb,       // [P, D]
    const int*   __restrict__ topk,     // [B, TOPK]
    unsigned short* __restrict__ ctx_bf,// [B, D]
    float* __restrict__ attn_out,       // [B, TOPK]
    int D) {
  const int b   = blockIdx.x;
  const int tid = threadIdx.x;
  const int h   = tid >> 4;     // head 0..15
  const int l   = tid & 15;     // lane in head

  __shared__ int   sidx[TOPK];
  __shared__ float sw[NHEADS][TOPK];
  if (tid < TOPK) sidx[tid] = topk[(size_t)b * TOPK + tid];
  __syncthreads();

  int idx[TOPK];
#pragma unroll
  for (int j = 0; j < TOPK; ++j) idx[j] = sidx[j];

  const int d0 = h * DHEAD + l * 4;
  const float4 qv = *(const float4*)(q + (size_t)b * D + d0);

  float sc[TOPK];
#pragma unroll
  for (int j = 0; j < TOPK; ++j) {
    const float4 kv = *(const float4*)(Kb + (size_t)idx[j] * D + d0);
    sc[j] = qv.x * kv.x + qv.y * kv.y + qv.z * kv.z + qv.w * kv.w;
  }
  // reduce across the 16 threads of this head (xor < 16 stays in-half, wave32)
#pragma unroll
  for (int m = 1; m < 16; m <<= 1)
#pragma unroll
    for (int j = 0; j < TOPK; ++j) sc[j] += __shfl_xor(sc[j], m, 32);

  // softmax (scale = 1/sqrt(64) = 0.125)
  float mx = -__builtin_inff();
#pragma unroll
  for (int j = 0; j < TOPK; ++j) {
    sc[j] *= 0.125f;
    mx = fmaxf(mx, sc[j]);
  }
  float e[TOPK], s = 0.0f;
#pragma unroll
  for (int j = 0; j < TOPK; ++j) {
    e[j] = __expf(sc[j] - mx);
    s += e[j];
  }
  const float inv = 1.0f / s;
  float w[TOPK];
#pragma unroll
  for (int j = 0; j < TOPK; ++j) w[j] = e[j] * inv;

  // weighted value sum
  float c0 = 0.f, c1 = 0.f, c2 = 0.f, c3 = 0.f;
#pragma unroll
  for (int j = 0; j < TOPK; ++j) {
    const float4 vv = *(const float4*)(Vb + (size_t)idx[j] * D + d0);
    c0 += w[j] * vv.x;
    c1 += w[j] * vv.y;
    c2 += w[j] * vv.z;
    c3 += w[j] * vv.w;
  }
  unsigned short* cp = ctx_bf + (size_t)b * D + d0;
  cp[0] = f32_to_bf16_rne(c0);
  cp[1] = f32_to_bf16_rne(c1);
  cp[2] = f32_to_bf16_rne(c2);
  cp[3] = f32_to_bf16_rne(c3);

  if (l == 0) {
#pragma unroll
    for (int j = 0; j < TOPK; ++j) sw[h][j] = w[j];
  }
  __syncthreads();
  if (tid < TOPK) {
    float acc = 0.0f;
    for (int hh = 0; hh < NHEADS; ++hh) acc += sw[hh][tid];
    attn_out[(size_t)b * TOPK + tid] = acc * (1.0f / NHEADS);
  }
}

// ---------------------------------------------------------------------------
// Host-side orchestration
// ---------------------------------------------------------------------------
extern "C" void kernel_launch(void* const* d_in, const int* in_sizes, int n_in,
                              void* d_out, int out_size, void* d_ws, size_t ws_size,
                              hipStream_t stream) {
  const float* query   = (const float*)d_in[0];
  const float* bank    = (const float*)d_in[1];
  const float* logits  = (const float*)d_in[2];
  const float* Wq_proj = (const float*)d_in[3];
  const float* bq_proj = (const float*)d_in[4];
  const float* Wp_proj = (const float*)d_in[5];
  const float* bp_proj = (const float*)d_in[6];
  const float* Wq      = (const float*)d_in[7];
  const float* bq      = (const float*)d_in[8];
  const float* Wk      = (const float*)d_in[9];
  const float* bk      = (const float*)d_in[10];
  const float* Wv      = (const float*)d_in[11];
  const float* bv      = (const float*)d_in[12];
  const float* Wo      = (const float*)d_in[13];
  const float* bo      = (const float*)d_in[14];

  const int D = 1024;
  const int B = in_sizes[0] / D;
  const int P = in_sizes[1] / D;

  // workspace carve-out (256B aligned)
  char* ws = (char*)d_ws;
  size_t off = 0;
  auto carve = [&](size_t bytes) -> void* {
    void* p = ws + off;
    off = (off + bytes + 255) & ~(size_t)255;
    return p;
  };
  unsigned short* query_bf = (unsigned short*)carve((size_t)B * D * 2);
  unsigned short* bank_bf  = (unsigned short*)carve((size_t)P * D * 2);
  unsigned short* Wqp_bf   = (unsigned short*)carve((size_t)D * D * 2);
  unsigned short* Wpp_bf   = (unsigned short*)carve((size_t)D * D * 2);
  unsigned short* Wq_bf    = (unsigned short*)carve((size_t)D * D * 2);
  unsigned short* Wk_bf    = (unsigned short*)carve((size_t)D * D * 2);
  unsigned short* Wv_bf    = (unsigned short*)carve((size_t)D * D * 2);
  unsigned short* Wo_bf    = (unsigned short*)carve((size_t)D * D * 2);
  unsigned short* qproj_bf = (unsigned short*)carve((size_t)B * D * 2);
  unsigned short* bankP_bf = (unsigned short*)carve((size_t)P * D * 2);
  float*          q_f      = (float*)carve((size_t)B * D * 4);
  float*          bankK_f  = (float*)carve((size_t)P * D * 4);
  float*          bankV_f  = (float*)carve((size_t)P * D * 4);
  unsigned short* ctx_bf   = (unsigned short*)carve((size_t)B * D * 2);
  int*            topk_idx = (int*)carve((size_t)B * TOPK * 4);

  float* out_ctx  = (float*)d_out;                 // [B, D]
  float* out_attn = out_ctx + (size_t)B * D;       // [B, TOPK]

  const int CVT_GRID = 1024;
  f32_to_bf16_kernel<<<CVT_GRID, 256, 0, stream>>>(query,   query_bf, B * D);
  f32_to_bf16_kernel<<<CVT_GRID, 256, 0, stream>>>(bank,    bank_bf,  P * D);
  f32_to_bf16_kernel<<<CVT_GRID, 256, 0, stream>>>(Wq_proj, Wqp_bf,   D * D);
  f32_to_bf16_kernel<<<CVT_GRID, 256, 0, stream>>>(Wp_proj, Wpp_bf,   D * D);
  f32_to_bf16_kernel<<<CVT_GRID, 256, 0, stream>>>(Wq,      Wq_bf,    D * D);
  f32_to_bf16_kernel<<<CVT_GRID, 256, 0, stream>>>(Wk,      Wk_bf,    D * D);
  f32_to_bf16_kernel<<<CVT_GRID, 256, 0, stream>>>(Wv,      Wv_bf,    D * D);
  f32_to_bf16_kernel<<<CVT_GRID, 256, 0, stream>>>(Wo,      Wo_bf,    D * D);

  // top-k of the logits (independent of GEMM chain)
  topk6_kernel<<<B, 256, 0, stream>>>(logits, topk_idx, P);

  // qproj = query @ Wq_proj^T + bq_proj  -> bf16
  gemm_bf16_wmma<<<dim3(D / 128, B / 128), 256, 0, stream>>>(
      (const __bf16*)query_bf, (const __bf16*)Wqp_bf, bq_proj,
      nullptr, qproj_bf, B, D, D);
  // bankP = bank @ Wp_proj^T + bp_proj  -> bf16
  gemm_bf16_wmma<<<dim3(D / 128, P / 128), 256, 0, stream>>>(
      (const __bf16*)bank_bf, (const __bf16*)Wpp_bf, bp_proj,
      nullptr, bankP_bf, P, D, D);
  // q = qproj @ Wq^T + bq  -> fp32
  gemm_bf16_wmma<<<dim3(D / 128, B / 128), 256, 0, stream>>>(
      (const __bf16*)qproj_bf, (const __bf16*)Wq_bf, bq,
      q_f, nullptr, B, D, D);
  // bankK = bankP @ Wk^T + bk  -> fp32  (project-then-gather: 12x fewer FLOPs)
  gemm_bf16_wmma<<<dim3(D / 128, P / 128), 256, 0, stream>>>(
      (const __bf16*)bankP_bf, (const __bf16*)Wk_bf, bk,
      bankK_f, nullptr, P, D, D);
  // bankV = bankP @ Wv^T + bv  -> fp32
  gemm_bf16_wmma<<<dim3(D / 128, P / 128), 256, 0, stream>>>(
      (const __bf16*)bankP_bf, (const __bf16*)Wv_bf, bv,
      bankV_f, nullptr, P, D, D);

  // attention over gathered top-k rows; emits ctx (bf16) + mean attn weights
  attn_topk6_kernel<<<B, 256, 0, stream>>>(q_f, bankK_f, bankV_f, topk_idx,
                                           ctx_bf, out_attn, D);

  // context = ctx @ Wo^T + bo  -> fp32, straight into d_out
  gemm_bf16_wmma<<<dim3(D / 128, B / 128), 256, 0, stream>>>(
      (const __bf16*)ctx_bf, (const __bf16*)Wo_bf, bo,
      out_ctx, nullptr, B, D, D);
}
